// WMG_Network_S_33285996544394
// MI455X (gfx1250) — compile-verified, matrix operand order checked by reading, not run on hardware
//
#include <hip/hip_runtime.h>
#include <hip/hip_bf16.h>
#include <math.h>

// ---------------------------------------------------------------------------
// WMG transformer forward on gfx1250 (CDNA5, wave32, WMMA).
// - All GEMMs ride v_wmma_f32_16x16x32_f16; f32 operands converted in-register.
// - Weights (576 MB f32, the HBM-bound stream) go direct global->VGPR as b128.
// - Activation A tiles are staged LDS-side via global_load_async_to_lds_b128
//   (ASYNCcnt) when the toolchain exposes the builtins; fallback is direct.
// - V is transposed per-head once per layer so the attn@V GEMM uses the same
//   contiguous mode as the weight GEMMs (no strided column loads).
// ---------------------------------------------------------------------------

typedef __attribute__((ext_vector_type(16))) _Float16 v16h;
typedef __attribute__((ext_vector_type(8)))  float    v8f;
typedef __attribute__((__vector_size__(16))) int      v4i_t;

#define Dm   1024
#define HIDm 4096
#define Lm   12
#define SV   257     // valid sequence length (1 + 256 concepts)
#define SP   288     // padded sequence length: 18*16 M-tiles, 9*32 N/K-tiles
#define HEADSm 16
#define DHm  64

#if __has_builtin(__builtin_amdgcn_global_load_async_to_lds_b128) && \
    __has_builtin(__builtin_amdgcn_s_wait_asynccnt)
#define USE_ASYNC_LDS 1
#else
#define USE_ASYNC_LDS 0
#endif

// ---------------------------------------------------------------------------
// GEMM:  C[b][m,n] = act( sum_k A[b][m,k] * B[b][n,k] + bias[n] )
// B is [N x K] row-major for every call (weights, K^T view, transposed V).
// One wave computes a 16x32 C region (two 16x16 tiles, shared A fragment);
// K stepped by 32 -> two WMMAs per step, four per unrolled body.
// ---------------------------------------------------------------------------
__global__ void gemm_wmma(const float* __restrict__ A, int lda, long strideA,
                          const float* __restrict__ B, int ldb, long strideB,
                          const float* __restrict__ bias,
                          float* __restrict__ C, int ldc, long strideC,
                          int M, int N, int K, int act)
{
#if USE_ASYNC_LDS
    __shared__ float aStage[8][2][16][32];   // [wave][buf][row][k] : 32 KB
#endif
    const int batch = blockIdx.y;
    const float* Ab = A + (size_t)batch * strideA;
    const float* Bb = B + (size_t)batch * strideB;
    float*       Cb = C + (size_t)batch * strideC;

    const int tilesN = N >> 5;                     // 32-wide N tiles
    const int tiles  = (M >> 4) * tilesN;
    const int wave   = (blockIdx.x * blockDim.x + threadIdx.x) >> 5;
    if (wave >= tiles) return;
    const int wib  = threadIdx.x >> 5;             // wave index in block (0..7)
    const int lane = threadIdx.x & 31;
    const int tm   = (wave / tilesN) << 4;
    const int tn   = (wave % tilesN) << 5;

    const int hs  = lane >> 4;                     // K half (0/1) this lane holds
    const int l16 = lane & 15;
#if USE_ASYNC_LDS
    const int cprow = lane >> 1;                   // staging copy: row 0..15
    const int cpoff = (lane & 1) * 4;              // staging copy: float 0 or 4
#endif

    v8f acc0 = {}, acc1 = {};
#pragma unroll 2
    for (int k0 = 0, it = 0; k0 < K; k0 += 32, ++it) {
        v16h a, b0, b1;

#if USE_ASYNC_LDS
        // ---- stage the 16x32 f32 A slab into LDS via the async engine
        {
            float* lbase = &aStage[wib][it & 1][cprow][cpoff];
            const float* gbase = Ab + (size_t)(tm + cprow) * lda + (k0 + cpoff);
#pragma unroll
            for (int j = 0; j < 4; ++j)
                __builtin_amdgcn_global_load_async_to_lds_b128(
                    (__attribute__((address_space(1))) v4i_t*)(gbase + j * 8),
                    (__attribute__((address_space(3))) v4i_t*)(lbase + j * 8),
                    0, 0);
            __builtin_amdgcn_s_wait_asynccnt(0);
            const float* lp = &aStage[wib][it & 1][l16][hs * 16];
#pragma unroll
            for (int j = 0; j < 16; ++j) a[j] = (_Float16)lp[j];
        }
#else
        // ---- A fragment direct: row (tm+l16), 16 consecutive K values
        {
            const float* ap = Ab + (size_t)(tm + l16) * lda + (k0 + 16 * hs);
#pragma unroll
            for (int j = 0; j < 16; ++j) a[j] = (_Float16)ap[j];
        }
#endif
        // ---- two B fragments (rows tn+l16 and tn+16+l16), contiguous K runs
        const float* bp0 = Bb + (size_t)(tn + l16) * ldb + (k0 + 16 * hs);
        const float* bp1 = bp0 + (size_t)16 * ldb;
#pragma unroll
        for (int j = 0; j < 16; ++j) b0[j] = (_Float16)bp0[j];
#pragma unroll
        for (int j = 0; j < 16; ++j) b1[j] = (_Float16)bp1[j];
        if (k0 + 32 < K) {                          // global_prefetch_b8
            __builtin_prefetch(bp0 + 32, 0, 1);
            __builtin_prefetch(bp1 + 32, 0, 1);
        }

        acc0 = __builtin_amdgcn_wmma_f32_16x16x32_f16(
                   false, a, false, b0, (short)0, acc0, false, false);
        acc1 = __builtin_amdgcn_wmma_f32_16x16x32_f16(
                   false, a, false, b1, (short)0, acc1, false, false);
    }

    // ---- writeback: VGPR r -> M = r (lanes 0-15) / r+8 (lanes 16-31)
    const int col0 = tn + l16;
    const int col1 = col0 + 16;
    const float bv0 = bias ? bias[col0] : 0.0f;
    const float bv1 = bias ? bias[col1] : 0.0f;
#pragma unroll
    for (int r = 0; r < 8; ++r) {
        const int row = tm + r + 8 * hs;
        float v0 = acc0[r] + bv0;
        float v1 = acc1[r] + bv1;
        if (act == 1) { v0 = fmaxf(v0, 0.0f); v1 = fmaxf(v1, 0.0f); }
        Cb[(size_t)row * ldc + col0] = v0;
        Cb[(size_t)row * ldc + col1] = v1;
    }
}

// ---------------------------------------------------------------------------
// Embedding: row 0 = W_core @ obs + b_core
//            row 1+i = W_cemb @ concept_i + b_cemb + W_age[:,i]
//                      + (i/256)*W_age[:,257] + b_age   (one-hot age trick)
//            rows >= 257 zeroed (padding).
// ---------------------------------------------------------------------------
__global__ void embed_kernel(const float* __restrict__ obs,
                             const float* __restrict__ concepts,
                             const float* __restrict__ Wc, const float* __restrict__ bc,
                             const float* __restrict__ We, const float* __restrict__ be,
                             const float* __restrict__ Wa, const float* __restrict__ ba,
                             float* __restrict__ x)
{
    const int d   = blockIdx.y * blockDim.x + threadIdx.x;   // 0..1023
    const int row = blockIdx.x;                              // 0..SP-1
    float acc;
    if (row >= SV) { x[(size_t)row * Dm + d] = 0.0f; return; }
    if (row == 0) {
        acc = bc[d];
        const float* w = Wc + (size_t)d * 2048;
        for (int k = 0; k < 2048; ++k) acc = fmaf(w[k], obs[k], acc);
    } else {
        const int i = row - 1;
        acc = be[d] + ba[d];
        const float* w = We + (size_t)d * 128;
        const float* c = concepts + (size_t)i * 128;
        for (int k = 0; k < 128; ++k) acc = fmaf(w[k], c[k], acc);
        const float* wa = Wa + (size_t)d * 258;
        acc += wa[i] + ((float)i * (1.0f / 256.0f)) * wa[257];
    }
    x[(size_t)row * Dm + d] = acc;
}

// ---------------------------------------------------------------------------
// Transpose V into per-head [DHm x SP] (N x K) so attn@V is a contiguous GEMM.
// vT[h][n][k] = v[k][h*64+n].  Grid (SP, Dm/256), 256 threads.
// ---------------------------------------------------------------------------
__global__ void transpose_v_kernel(const float* __restrict__ v,
                                   float* __restrict__ vT)
{
    const int k = blockIdx.x;                                // seq position
    const int d = blockIdx.y * blockDim.x + threadIdx.x;     // 0..1023
    const int h = d >> 6;
    const int n = d & 63;
    vT[((size_t)h * DHm + n) * SP + k] = v[(size_t)k * Dm + d];
}

// ---------------------------------------------------------------------------
// Row-wise softmax over valid 257 cols with 1/sqrt(64) scaling.
// Padded cols/rows zeroed so the attn@V GEMM can run K=288 cleanly.
// ---------------------------------------------------------------------------
__global__ void softmax_kernel(float* __restrict__ scores)
{
    const int head = blockIdx.x / SP;
    const int row  = blockIdx.x % SP;
    float* p = scores + ((size_t)head * SP + row) * SP;
    __shared__ float red[256];

    if (row >= SV) {
        for (int i = threadIdx.x; i < SP; i += 256) p[i] = 0.0f;
        return;
    }
    float m = -1e30f;
    for (int i = threadIdx.x; i < SV; i += 256) m = fmaxf(m, p[i] * 0.125f);
    red[threadIdx.x] = m; __syncthreads();
    for (int s = 128; s > 0; s >>= 1) {
        if (threadIdx.x < s) red[threadIdx.x] = fmaxf(red[threadIdx.x], red[threadIdx.x + s]);
        __syncthreads();
    }
    m = red[0]; __syncthreads();

    float sum = 0.0f;
    for (int i = threadIdx.x; i < SV; i += 256) {
        float e = __expf(p[i] * 0.125f - m);
        p[i] = e; sum += e;
    }
    red[threadIdx.x] = sum; __syncthreads();
    for (int s = 128; s > 0; s >>= 1) {
        if (threadIdx.x < s) red[threadIdx.x] += red[threadIdx.x + s];
        __syncthreads();
    }
    const float inv = 1.0f / red[0];
    for (int i = threadIdx.x; i < SV; i += 256) p[i] *= inv;
    for (int i = SV + threadIdx.x; i < SP; i += 256) p[i] = 0.0f;
}

// ---------------------------------------------------------------------------
// x[row] = LayerNorm(x[row] + t[row]) * g + b   (in place on x)
// ---------------------------------------------------------------------------
__global__ void add_ln_kernel(const float* __restrict__ t,
                              const float* __restrict__ g,
                              const float* __restrict__ b,
                              float* __restrict__ x)
{
    const int row = blockIdx.x;
    float* xp = x + (size_t)row * Dm;
    if (row >= SV) {
#pragma unroll
        for (int j = 0; j < 4; ++j) xp[threadIdx.x * 4 + j] = 0.0f;
        return;
    }
    const float* tp = t + (size_t)row * Dm;
    float v[4]; float s = 0.0f, s2 = 0.0f;
#pragma unroll
    for (int j = 0; j < 4; ++j) {
        const int i = threadIdx.x * 4 + j;
        float u = xp[i] + tp[i];
        v[j] = u; s += u; s2 = fmaf(u, u, s2);
    }
    __shared__ float r1[256], r2[256];
    r1[threadIdx.x] = s; r2[threadIdx.x] = s2; __syncthreads();
    for (int st = 128; st > 0; st >>= 1) {
        if (threadIdx.x < st) {
            r1[threadIdx.x] += r1[threadIdx.x + st];
            r2[threadIdx.x] += r2[threadIdx.x + st];
        }
        __syncthreads();
    }
    const float mean = r1[0] * (1.0f / Dm);
    const float var  = r2[0] * (1.0f / Dm) - mean * mean;
    const float inv  = rsqrtf(var + 1e-5f);
#pragma unroll
    for (int j = 0; j < 4; ++j) {
        const int i = threadIdx.x * 4 + j;
        xp[i] = (v[j] - mean) * inv * g[i] + b[i];
    }
}

// ---------------------------------------------------------------------------
// Output heads from token 0.
// ---------------------------------------------------------------------------
__global__ void head_kernel(const float* __restrict__ h,
                            const float* __restrict__ Wout, const float* __restrict__ bout,
                            const float* __restrict__ Wcon, const float* __restrict__ bcon,
                            float* __restrict__ out)
{
    const int i = blockIdx.x * blockDim.x + threadIdx.x;  // 0..639
    if (i < 512) {
        float acc = bout[i];
        const float* w = Wout + (size_t)i * Dm;
        for (int k = 0; k < Dm; ++k) acc = fmaf(w[k], h[k], acc);
        out[i] = fmaxf(acc, 0.0f);
    } else if (i < 640) {
        const int j = i - 512;
        float acc = bcon[j];
        const float* w = Wcon + (size_t)j * Dm;
        for (int k = 0; k < Dm; ++k) acc = fmaf(w[k], h[k], acc);
        out[i] = tanhf(acc);
    }
}

// ---------------------------------------------------------------------------
// Host-side launch orchestration
// ---------------------------------------------------------------------------
static inline void launch_gemm(hipStream_t st,
                               const float* A, int lda, long sA,
                               const float* B, int ldb, long sB,
                               const float* bias,
                               float* C, int ldc, long sC,
                               int M, int N, int K, int act, int batch)
{
    const int tiles = (M / 16) * (N / 32);
    dim3 grid((tiles + 7) / 8, batch);
    gemm_wmma<<<grid, 256, 0, st>>>(A, lda, sA, B, ldb, sB, bias,
                                    C, ldc, sC, M, N, K, act);
}

extern "C" void kernel_launch(void* const* d_in, const int* in_sizes, int n_in,
                              void* d_out, int out_size, void* d_ws, size_t ws_size,
                              hipStream_t stream)
{
    (void)in_sizes; (void)n_in; (void)out_size; (void)ws_size;

    const float* observation = (const float*)d_in[0];
    const float* concepts    = (const float*)d_in[1];
    const float* W_core = (const float*)d_in[2];  const float* b_core = (const float*)d_in[3];
    const float* W_cemb = (const float*)d_in[4];  const float* b_cemb = (const float*)d_in[5];
    const float* W_age  = (const float*)d_in[6];  const float* b_age  = (const float*)d_in[7];
    const float* Wq = (const float*)d_in[8];   const float* bq = (const float*)d_in[9];
    const float* Wk = (const float*)d_in[10];  const float* bk = (const float*)d_in[11];
    const float* Wv = (const float*)d_in[12];  const float* bv = (const float*)d_in[13];
    const float* Wo = (const float*)d_in[14];  const float* bo = (const float*)d_in[15];
    const float* ln1_g = (const float*)d_in[16]; const float* ln1_b = (const float*)d_in[17];
    const float* W1 = (const float*)d_in[18];  const float* b1 = (const float*)d_in[19];
    const float* W2 = (const float*)d_in[20];  const float* b2 = (const float*)d_in[21];
    const float* ln2_g = (const float*)d_in[22]; const float* ln2_b = (const float*)d_in[23];
    const float* W_concept = (const float*)d_in[24]; const float* b_concept = (const float*)d_in[25];
    const float* W_out     = (const float*)d_in[26]; const float* b_out     = (const float*)d_in[27];
    float* out = (float*)d_out;

    // -------- workspace carve-up (floats) --------
    float* ws = (float*)d_ws;
    const size_t SD  = (size_t)SP * Dm;            // 288*1024
    float* x        = ws;  ws += SD;               // activations (in-place updated)
    float* tmp      = ws;  ws += SD;               // projection result (pre-LN residual)
    float* q        = ws;  ws += SD;
    float* kbuf     = ws;  ws += SD;
    float* vbuf     = ws;  ws += SD;
    float* vT       = ws;  ws += SD;               // per-head transposed V [16][64][288]
    float* attn_out = ws;  ws += SD;
    float* scores   = ws;  ws += (size_t)HEADSm * SP * SP;   // 16*288*288
    float* ffn1     = ws;  ws += (size_t)SP * HIDm;          // 288*4096

    // -------- embedding --------
    {
        dim3 grid(SP, Dm / 256);
        embed_kernel<<<grid, 256, 0, stream>>>(observation, concepts,
                                               W_core, b_core, W_cemb, b_cemb,
                                               W_age, b_age, x);
    }

    // -------- transformer layers --------
    const long WS  = (long)Dm * Dm;
    const long W1S = (long)HIDm * Dm;
    const long W2S = (long)Dm * HIDm;

    for (int l = 0; l < Lm; ++l) {
        const float* Wq_l = Wq + (size_t)l * WS;  const float* bq_l = bq + (size_t)l * Dm;
        const float* Wk_l = Wk + (size_t)l * WS;  const float* bk_l = bk + (size_t)l * Dm;
        const float* Wv_l = Wv + (size_t)l * WS;  const float* bv_l = bv + (size_t)l * Dm;
        const float* Wo_l = Wo + (size_t)l * WS;  const float* bo_l = bo + (size_t)l * Dm;
        const float* W1_l = W1 + (size_t)l * W1S; const float* b1_l = b1 + (size_t)l * HIDm;
        const float* W2_l = W2 + (size_t)l * W2S; const float* b2_l = b2 + (size_t)l * Dm;

        // Q/K/V projections: [288,1024] = x @ W^T + b
        launch_gemm(stream, x, Dm, 0, Wq_l, Dm, 0, bq_l, q,    Dm, 0, SP, Dm, Dm, 0, 1);
        launch_gemm(stream, x, Dm, 0, Wk_l, Dm, 0, bk_l, kbuf, Dm, 0, SP, Dm, Dm, 0, 1);
        launch_gemm(stream, x, Dm, 0, Wv_l, Dm, 0, bv_l, vbuf, Dm, 0, SP, Dm, Dm, 0, 1);

        // per-head V transpose -> [h][64][288]
        transpose_v_kernel<<<dim3(SP, Dm / 256), 256, 0, stream>>>(vbuf, vT);

        // scores[h] = Q_h @ K_h^T   (batched over 16 heads, K-dim = 64)
        launch_gemm(stream, q, Dm, DHm, kbuf, Dm, DHm, nullptr,
                    scores, SP, (long)SP * SP, SP, SP, DHm, 0, HEADSm);

        // softmax (scale 1/8, zero padding)
        softmax_kernel<<<dim3(HEADSm * SP), 256, 0, stream>>>(scores);

        // attn_out[h] = P_h @ V_h   via transposed V (contiguous K runs)
        launch_gemm(stream, scores, SP, (long)SP * SP, vT, SP, (long)DHm * SP, nullptr,
                    attn_out, Dm, DHm, SP, DHm, SP, 0, HEADSm);

        // output projection + residual + LN1
        launch_gemm(stream, attn_out, Dm, 0, Wo_l, Dm, 0, bo_l, tmp, Dm, 0, SP, Dm, Dm, 0, 1);
        add_ln_kernel<<<dim3(SP), 256, 0, stream>>>(tmp, ln1_g + (size_t)l * Dm,
                                                    ln1_b + (size_t)l * Dm, x);

        // FFN: relu(x @ W1^T + b1) @ W2^T + b2, residual + LN2
        launch_gemm(stream, x, Dm, 0, W1_l, Dm, 0, b1_l, ffn1, HIDm, 0, SP, HIDm, Dm, 1, 1);
        launch_gemm(stream, ffn1, HIDm, 0, W2_l, HIDm, 0, b2_l, tmp, Dm, 0, SP, Dm, HIDm, 0, 1);
        add_ln_kernel<<<dim3(SP), 256, 0, stream>>>(tmp, ln2_g + (size_t)l * Dm,
                                                    ln2_b + (size_t)l * Dm, x);
    }

    // -------- output heads from token 0 --------
    head_kernel<<<dim3(5), 128, 0, stream>>>(x, W_out, b_out, W_concept, b_concept, out);
}
